// HyperGraphResBlock_23476291240117
// MI455X (gfx1250) — matrix operand order repaired and checked by reading, not compiled, not run on hardware
//
#include <hip/hip_runtime.h>
#include <stdint.h>

// ---- problem constants (match reference) ----
#define B_    64
#define N_    1723
#define E_    1723
#define NNZ_  16384
#define CIN_  256
#define COUT_ 256
#define H1_   64
#define H2_   32
#define H3_   128
#define RTOT  (N_ * B_)        // 110272 = 16 * 6892
#define MT    (RTOT / 16)      // 6892 M-tiles

typedef __attribute__((ext_vector_type(16))) _Float16 v16h;
typedef __attribute__((ext_vector_type(8)))  float    v8f;

// ---------------- wave reductions (wave32) ----------------
__device__ __forceinline__ float wred32(float v) {
#pragma unroll
  for (int m = 16; m >= 1; m >>= 1) v += __shfl_xor(v, m, 32);
  return v;
}
__device__ __forceinline__ float wred8(float v) {
  v += __shfl_xor(v, 4, 32);
  v += __shfl_xor(v, 2, 32);
  v += __shfl_xor(v, 1, 32);
  return v;
}

// ---------------- CDNA5 async memory->LDS copy (ASYNCcnt path) ----------------
// 16B per lane: LDS byte offset in a VGPR, 64-bit global address in a VGPR pair.
__device__ __forceinline__ void async_b128(unsigned lds_off, const void* g) {
  asm volatile("global_load_async_to_lds_b128 %0, %1, off"
               :: "v"(lds_off), "v"((unsigned long long)(uintptr_t)g)
               : "memory");
}
__device__ __forceinline__ void async_wait0() {
  asm volatile("s_wait_asynccnt 0" ::: "memory");
}
// low 32 bits of the flat address of a __shared__ object == LDS byte offset
__device__ __forceinline__ unsigned lds_off32(const void* p) {
  return (unsigned)(uintptr_t)p;
}

// ---------------- WMMA fragment builders ----------------
// A (16xK f16, row-major tile in LDS). k order used here (contiguous, same as frag_b_h):
// lane L (L<16): row M=L, K0..K0+15; lane L+16: K0+16..K0+31.
__device__ __forceinline__ v16h frag_a_lds(const _Float16* tile, int ld, int k0, int lane) {
  int m  = lane & 15;
  int kh = (lane >> 4) << 4;         // 0 or 16
  const _Float16* p = tile + m * ld + k0 + kh;
  v16h a;
#pragma unroll
  for (int i = 0; i < 8; ++i) a[i]     = p[i];
#pragma unroll
  for (int i = 0; i < 8; ++i) a[8 + i] = p[8 + i];
  return a;
}
// same, but tile kept in LDS as fp32 (converted while building the fragment)
__device__ __forceinline__ v16h frag_a_lds_f32(const float* tile, int ld, int k0, int lane) {
  int m  = lane & 15;
  int kh = (lane >> 4) << 4;
  const float* p = tile + m * ld + k0 + kh;
  v16h a;
#pragma unroll
  for (int i = 0; i < 8; ++i) a[i]     = (_Float16)p[i];
#pragma unroll
  for (int i = 0; i < 8; ++i) a[8 + i] = (_Float16)p[8 + i];
  return a;
}
// B (32x16 f16) from pre-converted f16 weight Wh[O][K] (Y = A * Wh^T):
// lane l: col = l&15 (output channel), k = (l>>4)*16 + i -> 32B contiguous.
__device__ __forceinline__ v16h frag_b_h(const _Float16* Wh, int ldk, int o0, int k0, int lane) {
  int col = lane & 15;
  int kh  = (lane >> 4) << 4;        // 0 or 16
  return *(const v16h*)(Wh + (size_t)(o0 + col) * ldk + k0 + kh);
}

// ---------------- weight fp32 -> f16 pre-convert ----------------
__global__ void k_cvt_w(const float* __restrict__ a, _Float16* __restrict__ o, int n) {
  int i = blockIdx.x * 256 + threadIdx.x;
  if (i < n) o[i] = (_Float16)a[i];
}

// ---------------- K1: LN(x)+ReLU -> f16, (B,N,C) -> (N,B,C) ----------------
__global__ void k_ln_relu_cast(const float* __restrict__ x,
                               const float* __restrict__ g, const float* __restrict__ bb,
                               _Float16* __restrict__ A1) {
  int lane = threadIdx.x & 31;
  int w    = threadIdx.x >> 5;
  int rbn  = blockIdx.x * 8 + w;        // b*N + n
  int b = rbn / N_, n = rbn % N_;
  const float* px = x + (size_t)rbn * CIN_;
  float v[8], s = 0.f, s2 = 0.f;
#pragma unroll
  for (int i = 0; i < 8; ++i) { v[i] = px[i * 32 + lane]; s += v[i]; s2 += v[i] * v[i]; }
  s = wred32(s); s2 = wred32(s2);
  float mu  = s * (1.f / CIN_);
  float var = s2 * (1.f / CIN_) - mu * mu;
  float rs  = rsqrtf(var + 1e-5f);
  _Float16* pa = A1 + ((size_t)n * B_ + b) * CIN_;
#pragma unroll
  for (int i = 0; i < 8; ++i) {
    int c = i * 32 + lane;
    float t = (v[i] - mu) * rs * g[c] + bb[c];
    pa[c] = (_Float16)fmaxf(t, 0.f);
  }
}

// ---------------- K2: lin1 GEMM (K=256 -> 64) + bias + LN + ReLU -> f16 ----------------
__global__ void k_lin1_ln(const _Float16* __restrict__ A1,
                          const _Float16* __restrict__ Wh, const float* __restrict__ bias,
                          const float* __restrict__ g, const float* __restrict__ bb,
                          _Float16* __restrict__ A2) {
  __shared__ __align__(16) _Float16 a_lds[16 * CIN_];   // 8 KB
  __shared__ float y_lds[16 * H1_];                     // 4 KB
  int tid  = threadIdx.x;           // 128 threads = 4 waves
  int lane = tid & 31, w = tid >> 5;
  int r0   = blockIdx.x * 16;
  // async-stage full 16x256 f16 A tile (8 KB = 512 x b128), 4 chunks/thread
  {
    unsigned l0 = lds_off32(a_lds);
    const char* gb = (const char*)(A1 + (size_t)r0 * CIN_);
#pragma unroll
    for (int i = 0; i < 4; ++i) {
      unsigned off = (unsigned)(tid + i * 128) * 16u;
      async_b128(l0 + off, gb + off);
    }
    async_wait0();
  }
  __syncthreads();

  v8f acc = {};
#pragma unroll
  for (int kt = 0; kt < CIN_ / 32; ++kt) {
    v16h a   = frag_a_lds(a_lds, CIN_, kt * 32, lane);
    v16h bfr = frag_b_h(Wh, CIN_, w * 16, kt * 32, lane);
    acc = __builtin_amdgcn_wmma_f32_16x16x32_f16(false, a, false, bfr, (short)0, acc, false, false);
  }
  int col = w * 16 + (lane & 15);
  float bs = bias[col];
  int m0 = (lane >> 4) * 8;
#pragma unroll
  for (int j = 0; j < 8; ++j) y_lds[(m0 + j) * H1_ + col] = acc[j] + bs;
  __syncthreads();

  // LN over 64 channels: 16 rows x 8 threads
  int row = tid >> 3, sub = tid & 7;
  float v[8], s = 0.f, s2 = 0.f;
#pragma unroll
  for (int i = 0; i < 8; ++i) { v[i] = y_lds[row * H1_ + sub + i * 8]; s += v[i]; s2 += v[i] * v[i]; }
  s = wred8(s); s2 = wred8(s2);
  float mu = s * (1.f / H1_), var = s2 * (1.f / H1_) - mu * mu, rs = rsqrtf(var + 1e-5f);
  _Float16* pa = A2 + (size_t)(r0 + row) * H1_;
#pragma unroll
  for (int i = 0; i < 8; ++i) {
    int c = sub + i * 8;
    float t = (v[i] - mu) * rs * g[c] + bb[c];
    pa[c] = (_Float16)fmaxf(t, 0.f);
  }
}

// ---------------- generic theta GEMM: X(RxK) * Th(OxK)^T -> out(RxO) fp32 ----------------
// F16IN: input tile already f16; else fp32 tile staged raw and converted in frag build.
template <int O, int K, bool F16IN>
__global__ void k_theta_gemm(const void* __restrict__ Xv, const _Float16* __restrict__ Th,
                             float* __restrict__ out) {
  constexpr int ESZ     = F16IN ? 2 : 4;
  constexpr int TILE_B  = 16 * K * ESZ;       // bytes
  constexpr int CHUNKS  = TILE_B / 16;
  constexpr int NW      = O / 16;
  constexpr int NT      = NW * 32;
  __shared__ __align__(16) char a_raw[TILE_B];
  int tid = threadIdx.x;             // NW*32
  int lane = tid & 31, w = tid >> 5;
  int r0 = blockIdx.x * 16;
  {
    unsigned l0 = lds_off32(a_raw);
    const char* gb = (const char*)Xv + (size_t)r0 * K * ESZ;
    for (int c = tid; c < CHUNKS; c += NT) {
      unsigned off = (unsigned)c * 16u;
      async_b128(l0 + off, gb + off);
    }
    async_wait0();
  }
  __syncthreads();
  v8f acc = {};
#pragma unroll
  for (int kt = 0; kt < K / 32; ++kt) {
    v16h a;
    if constexpr (F16IN) a = frag_a_lds((const _Float16*)a_raw, K, kt * 32, lane);
    else                 a = frag_a_lds_f32((const float*)a_raw, K, kt * 32, lane);
    v16h bfr = frag_b_h(Th, K, w * 16, kt * 32, lane);
    acc = __builtin_amdgcn_wmma_f32_16x16x32_f16(false, a, false, bfr, (short)0, acc, false, false);
  }
  int col = w * 16 + (lane & 15);
  int m0 = (lane >> 4) * 8;
#pragma unroll
  for (int j = 0; j < 8; ++j) out[(size_t)(r0 + m0 + j) * O + col] = acc[j];
}

// ---------------- hypergraph helpers ----------------
__global__ void k_zero(float* p, int n) {
  int i = blockIdx.x * 256 + threadIdx.x;
  if (i < n) p[i] = 0.f;
}
__global__ void k_initb(float* p, const float* __restrict__ bias, int C, int n) {
  int i = blockIdx.x * 256 + threadIdx.x;
  if (i < n) p[i] = bias[i % C];
}
__global__ void k_count(const int* __restrict__ node, const int* __restrict__ edge,
                        float* Dd, float* Bd) {
  int i = blockIdx.x * 256 + threadIdx.x;
  if (i < NNZ_) {
    unsafeAtomicAdd(&Dd[node[i]], 1.f);
    unsafeAtomicAdd(&Bd[edge[i]], 1.f);
  }
}
__global__ void k_inv(float* p, int n) {
  int i = blockIdx.x * 256 + threadIdx.x;
  if (i < n) p[i] = (p[i] > 0.f) ? (1.f / p[i]) : 0.f;
}
// node -> hyperedge: ef[edge] += x[node] * Binv[edge]
template <int C>
__global__ void k_scatter1(const float* __restrict__ xl, float* __restrict__ ef,
                           const int* __restrict__ node, const int* __restrict__ edge,
                           const float* __restrict__ Binv) {
  constexpr int BC = B_ * C;
  constexpr int CH = BC / 256;
  int e = blockIdx.x / CH;
  int i = (blockIdx.x % CH) * 256 + threadIdx.x;
  int nd = node[e], ed = edge[e];
  float s = Binv[ed];
  unsafeAtomicAdd(&ef[(size_t)ed * BC + i], xl[(size_t)nd * BC + i] * s);
}
// hyperedge -> node: out[node] += ef[edge] * Dinv[node]
template <int C>
__global__ void k_scatter2(const float* __restrict__ ef, float* __restrict__ out,
                           const int* __restrict__ node, const int* __restrict__ edge,
                           const float* __restrict__ Dinv) {
  constexpr int BC = B_ * C;
  constexpr int CH = BC / 256;
  int e = blockIdx.x / CH;
  int i = (blockIdx.x % CH) * 256 + threadIdx.x;
  int nd = node[e], ed = edge[e];
  float s = Dinv[nd];
  unsafeAtomicAdd(&out[(size_t)nd * BC + i], ef[(size_t)ed * BC + i] * s);
}

// ---------------- final: LN(128)+ReLU -> lin2 GEMM (128->256) + bias + skip ----------------
__global__ void k_final(const float* __restrict__ H, const float* __restrict__ g,
                        const float* __restrict__ bb, const _Float16* __restrict__ Wh,
                        const float* __restrict__ bias, const float* __restrict__ x,
                        float* __restrict__ out) {
  __shared__ __align__(16) _Float16 a_lds[16 * H3_];    // 4 KB
  int tid = threadIdx.x;            // 512 threads = 16 waves
  int lane = tid & 31, w = tid >> 5;
  int r0 = blockIdx.x * 16;
  {  // LN phase: wave w normalizes row w (128 channels, 4 per lane)
    int r = r0 + w;
    const float* ph = H + (size_t)r * H3_;
    float v[4], s = 0.f, s2 = 0.f;
#pragma unroll
    for (int i = 0; i < 4; ++i) { v[i] = ph[i * 32 + lane]; s += v[i]; s2 += v[i] * v[i]; }
    s = wred32(s); s2 = wred32(s2);
    float mu = s * (1.f / H3_), var = s2 * (1.f / H3_) - mu * mu, rs = rsqrtf(var + 1e-5f);
#pragma unroll
    for (int i = 0; i < 4; ++i) {
      int c = i * 32 + lane;
      float t = (v[i] - mu) * rs * g[c] + bb[c];
      a_lds[w * H3_ + c] = (_Float16)fmaxf(t, 0.f);
    }
  }
  __syncthreads();
  v8f acc = {};
#pragma unroll
  for (int kt = 0; kt < H3_ / 32; ++kt) {
    v16h a   = frag_a_lds(a_lds, H3_, kt * 32, lane);
    v16h bfr = frag_b_h(Wh, H3_, w * 16, kt * 32, lane);
    acc = __builtin_amdgcn_wmma_f32_16x16x32_f16(false, a, false, bfr, (short)0, acc, false, false);
  }
  int col = w * 16 + (lane & 15);
  float bs = bias[col];
  int m0 = (lane >> 4) * 8;
#pragma unroll
  for (int j = 0; j < 8; ++j) {
    int r = r0 + m0 + j;        // r = n*B + b
    int n = r >> 6, b = r & 63;
    size_t idx = ((size_t)b * N_ + n) * COUT_ + col;
    out[idx] = acc[j] + bs + x[idx];
  }
}

// ---------------- launcher ----------------
extern "C" void kernel_launch(void* const* d_in, const int* in_sizes, int n_in,
                              void* d_out, int out_size, void* d_ws, size_t ws_size,
                              hipStream_t stream) {
  const float* x      = (const float*)d_in[0];
  const int*   inc    = (const int*)d_in[1];
  const float* pre_g  = (const float*)d_in[2];
  const float* pre_b  = (const float*)d_in[3];
  const float* lin1_W = (const float*)d_in[4];
  const float* lin1_b = (const float*)d_in[5];
  const float* n1_g   = (const float*)d_in[6];
  const float* n1_b   = (const float*)d_in[7];
  const float* th1    = (const float*)d_in[8];
  const float* bi1    = (const float*)d_in[9];
  const float* th2    = (const float*)d_in[10];
  const float* bi2    = (const float*)d_in[11];
  const float* n2_g   = (const float*)d_in[12];
  const float* n2_b   = (const float*)d_in[13];
  const float* lin2_W = (const float*)d_in[14];
  const float* lin2_b = (const float*)d_in[15];
  float* out = (float*)d_out;
  char*  ws  = (char*)d_ws;

  const int* node = inc;
  const int* edge = inc + NNZ_;

  const size_t R      = RTOT;
  const size_t sz_big = R * 128 * 4;          // 56,459,264 B (== R*256*2)

  // BIG0: A1 (f16) -> Xl2 (f32) -> H2 (f32), each consumer finishes before reuse
  _Float16* A1  = (_Float16*)(ws);
  float*    Xl2 = (float*)(ws);
  float*    H2  = (float*)(ws);
  // BIG1: early {A2, Xl1, ef1, H1} -> late ef2
  char* big1 = ws + sz_big;
  _Float16* A2  = (_Float16*)(big1);                                   // R*64 f16
  float*    Xl1 = (float*)(big1 + R * 64 * 2);                         // R*32 f32
  float*    ef1 = (float*)(big1 + R * 64 * 2 + R * 32 * 4);            // E*B*32 f32
  float*    H1  = (float*)(big1 + R * 64 * 2 + R * 32 * 4 + R * 32 * 4);
  float*    ef2 = (float*)(big1);                                      // E*B*128 f32
  // small tail: degree inverses + f16 weights
  char* tail = ws + 2 * sz_big;
  float* Dinv = (float*)tail;
  float* Binv = Dinv + N_;
  _Float16* Wh1 = (_Float16*)(tail + 16384);        // 64*256
  _Float16* Wt1 = Wh1 + (size_t)H1_ * CIN_;         // 32*64
  _Float16* Wt2 = Wt1 + (size_t)H2_ * H1_;          // 128*32
  _Float16* Wh2 = Wt2 + (size_t)H3_ * H2_;          // 256*128

  // weight pre-conversion (tiny, L2-resident afterwards)
  k_cvt_w<<<(H1_ * CIN_ + 255) / 256, 256, 0, stream>>>(lin1_W, Wh1, H1_ * CIN_);
  k_cvt_w<<<(H2_ * H1_ + 255) / 256, 256, 0, stream>>>(th1, Wt1, H2_ * H1_);
  k_cvt_w<<<(H3_ * H2_ + 255) / 256, 256, 0, stream>>>(th2, Wt2, H3_ * H2_);
  k_cvt_w<<<(COUT_ * H3_ + 255) / 256, 256, 0, stream>>>(lin2_W, Wh2, COUT_ * H3_);

  // degrees: count then invert (Dinv,Binv contiguous)
  k_zero<<<(N_ + E_ + 255) / 256, 256, 0, stream>>>(Dinv, N_ + E_);
  k_count<<<(NNZ_ + 255) / 256, 256, 0, stream>>>(node, edge, Dinv, Binv);
  k_inv<<<(N_ + E_ + 255) / 256, 256, 0, stream>>>(Dinv, N_ + E_);

  // pre-LN + ReLU + cast + (B,N,C)->(N,B,C)
  k_ln_relu_cast<<<RTOT / 8, 256, 0, stream>>>(x, pre_g, pre_b, A1);
  // lin1 + LN + ReLU (WMMA, async-LDS staging)
  k_lin1_ln<<<MT, 128, 0, stream>>>(A1, Wh1, lin1_b, n1_g, n1_b, A2);

  // hconv1: theta GEMM (WMMA), then two atomic scatter passes, bias folded into init
  k_theta_gemm<H2_, H1_, true><<<MT, (H2_ / 16) * 32, 0, stream>>>(A2, Wt1, Xl1);
  k_zero<<<(int)((R * 32 + 255) / 256), 256, 0, stream>>>(ef1, (int)(R * 32));
  k_initb<<<(int)((R * 32 + 255) / 256), 256, 0, stream>>>(H1, bi1, H2_, (int)(R * 32));
  k_scatter1<H2_><<<NNZ_ * (B_ * H2_ / 256), 256, 0, stream>>>(Xl1, ef1, node, edge, Binv);
  k_scatter2<H2_><<<NNZ_ * (B_ * H2_ / 256), 256, 0, stream>>>(ef1, H1, node, edge, Dinv);

  // hconv2
  k_theta_gemm<H3_, H2_, false><<<MT, (H3_ / 16) * 32, 0, stream>>>(H1, Wt2, Xl2);
  k_zero<<<(int)((R * 128 + 255) / 256), 256, 0, stream>>>(ef2, (int)(R * 128));
  k_scatter1<H3_><<<NNZ_ * (B_ * H3_ / 256), 256, 0, stream>>>(Xl2, ef2, node, edge, Binv);
  k_initb<<<(int)((R * 128 + 255) / 256), 256, 0, stream>>>(H2, bi2, H3_, (int)(R * 128));
  k_scatter2<H3_><<<NNZ_ * (B_ * H3_ / 256), 256, 0, stream>>>(ef2, H2, node, edge, Dinv);

  // post-LN + ReLU + lin2 (WMMA) + bias + skip-add, writing (B,N,256)
  k_final<<<MT, 512, 0, stream>>>(H2, n2_g, n2_b, lin2_W ? Wh2 : Wh2, lin2_b, x, out);
}
